// DGM_Model_48550310314079
// MI455X (gfx1250) — matrix operand order repaired and checked by reading, not compiled, not run on hardware
//
#include <hip/hip_runtime.h>
#include <hip/hip_bf16.h>

typedef _Float16 v16h __attribute__((ext_vector_type(16)));
typedef float    v8f  __attribute__((ext_vector_type(8)));
typedef unsigned long long u64;

#define NPB   4096      // nodes per batch
#define NB    2         // batches
#define NTOT  8192      // NB*NPB
#define KNN   5
#define NK    20480     // NPB*KNN
#define NEDGE 40960     // NK*NB
#define DH    64

// ---------------------------------------------------------------- helpers
__device__ __forceinline__ unsigned enc_f32(float f) {
  unsigned u = __float_as_uint(f);
  return u ^ ((unsigned)((int)u >> 31) | 0x80000000u);   // order-preserving, 3 VALU
}
__device__ __forceinline__ float dec_f32(unsigned e) {
  unsigned u = (e & 0x80000000u) ? (e & 0x7FFFFFFFu) : ~e;
  return __uint_as_float(u);
}

// Branch-free sorted top-5 (ascending) insert on 64-bit keys.
// Uses sortedness: new[m] = clamp(k, old[m-1], old[m]); pure v_min/max_u64 data flow.
__device__ __forceinline__ void top5_insert64(u64 (&b)[5], u64 k) {
#pragma unroll
  for (int m = 4; m >= 1; --m) {
    u64 lo = b[m - 1];
    u64 up = k > lo ? k : lo;          // max
    b[m]   = up < b[m] ? up : b[m];    // min
  }
  b[0] = k < b[0] ? k : b[0];
}

// ---------------------------------------------------------------- pre_fc: [N,32] -> leaky_relu -> [N,64]
__global__ __launch_bounds__(64) void k_prefc(const float* __restrict__ x,
                                              const float* __restrict__ W,
                                              const float* __restrict__ b,
                                              float* __restrict__ xf) {
  int row = blockIdx.x, t = threadIdx.x;
  __shared__ float sx[32];
  if (t < 32) sx[t] = x[(size_t)row * 32 + t];
  __syncthreads();
  float acc = b[t];
#pragma unroll 8
  for (int c = 0; c < 32; ++c) acc += sx[c] * W[c * 64 + t];
  acc = acc > 0.0f ? acc : 0.1f * acc;
  xf[(size_t)row * 64 + t] = acc;
}

// ---------------------------------------------------------------- embed: [a | a2?] @ W + b -> ge (f32 + f16)
__global__ __launch_bounds__(64) void k_embed(const float* __restrict__ a,
                                              const float* __restrict__ a2,
                                              const float* __restrict__ W,
                                              const float* __restrict__ b,
                                              float* __restrict__ ge,
                                              _Float16* __restrict__ geh) {
  int row = blockIdx.x, t = threadIdx.x;
  __shared__ float s[128];
  s[t] = a[(size_t)row * 64 + t];
  if (a2) s[64 + t] = a2[(size_t)row * 64 + t];
  __syncthreads();
  int cn = a2 ? 128 : 64;
  float acc = b[t];
#pragma unroll 8
  for (int c = 0; c < cn; ++c) acc += s[c] * W[c * 64 + t];
  ge[(size_t)row * 64 + t]  = acc;
  geh[(size_t)row * 64 + t] = (_Float16)acc;
}

// ---------------------------------------------------------------- per-row squared norm
__global__ void k_rowsq(const float* __restrict__ ge, float* __restrict__ sq) {
  int r = blockIdx.x * blockDim.x + threadIdx.x;
  if (r >= NTOT) return;
  const float4* p = (const float4*)(ge + (size_t)r * 64);
  float s = 0.f;
#pragma unroll
  for (int i = 0; i < 16; ++i) {
    float4 v = p[i];
    s += v.x * v.x + v.y * v.y + v.z * v.z + v.w * v.w;
  }
  sq[r] = s;
}

// ---------------------------------------------------------------- fused gram(WMMA) + column top-5
// grid (4096/64, NB), block 128 (4 waves); wave w handles 16 columns.
__global__ __launch_bounds__(128) void k_topk(const _Float16* __restrict__ geh,
                                              const float* __restrict__ sq,
                                              int* __restrict__ flat) {
  const int n = NPB;
  int lane = threadIdx.x & 31;
  int wave = threadIdx.x >> 5;
  int batch = blockIdx.y;
  int colbase = blockIdx.x * 64 + wave * 16;   // within batch
  int jn = lane & 15;
  int hi = lane >> 4;

  const _Float16* gb = geh + (size_t)batch * n * 64;
  const float*   sb = sq + batch * n;

  // B fragments (K x 16 columns): lane = column, table-order K (8h+s, 16+8h+s) — loaded once
  v16h b0, b1;
  {
    const _Float16* bp = gb + (size_t)(colbase + jn) * 64 + hi * 8;
    uint4* q0 = (uint4*)&b0;
    q0[0] = *(const uint4*)(bp + 0);
    q0[1] = *(const uint4*)(bp + 16);
    uint4* q1 = (uint4*)&b1;
    q1[0] = *(const uint4*)(bp + 32);
    q1[1] = *(const uint4*)(bp + 48);
  }

  u64 best[5];
#pragma unroll
  for (int m = 0; m < 5; ++m) best[m] = ~0ull;

  // software pipeline: prefetch A fragments for the next tile while current computes
  v16h a0n, a1n;
  {
    const _Float16* ap = gb + (size_t)jn * 64 + hi * 8;
    uint4* p0 = (uint4*)&a0n;
    p0[0] = *(const uint4*)(ap + 0);
    p0[1] = *(const uint4*)(ap + 16);
    uint4* p1 = (uint4*)&a1n;
    p1[0] = *(const uint4*)(ap + 32);
    p1[1] = *(const uint4*)(ap + 48);
  }

  for (int rowbase = 0; rowbase < n; rowbase += 16) {
    v16h a0 = a0n, a1 = a1n;

    // vectorized row-norm tile: 8 contiguous floats for this lane-half
    float4 s0 = *(const float4*)(sb + rowbase + hi * 8);
    float4 s1 = *(const float4*)(sb + rowbase + hi * 8 + 4);

    // prefetch next tile's A fragments (wrapped, branch-free)
    {
      int nr = (rowbase + 16) & (n - 1);
      const _Float16* ap = gb + (size_t)(nr + jn) * 64 + hi * 8;
      uint4* p0 = (uint4*)&a0n;
      p0[0] = *(const uint4*)(ap + 0);
      p0[1] = *(const uint4*)(ap + 16);
      uint4* p1 = (uint4*)&a1n;
      p1[0] = *(const uint4*)(ap + 32);
      p1[1] = *(const uint4*)(ap + 48);
    }

    v8f c = {};
    c = __builtin_amdgcn_wmma_f32_16x16x32_f16(false, a0, false, b0, (short)0, c, false, false);
    c = __builtin_amdgcn_wmma_f32_16x16x32_f16(false, a1, false, b1, (short)0, c, false, false);

    float sv[8] = {s0.x, s0.y, s0.z, s0.w, s1.x, s1.y, s1.z, s1.w};

    // C layout: lane<16 -> rows rowbase+v, lane>=16 -> rows rowbase+8+v, col = lane&15
#pragma unroll
    for (int v = 0; v < 8; ++v) {
      int i = rowbase + v + hi * 8;
      float key = sv[v] - 2.0f * c[v];       // order-equivalent to t*max(sqi+sqj-2g,0)
      u64 kk = ((u64)enc_f32(key) << 32) | (unsigned)i;  // ties -> lower index first
      if (kk < best[4]) top5_insert64(best, kk);         // usually skipped (execz)
    }
  }

  // merge the two half-lists per column via cross-lane shuffle (wave32)
#pragma unroll
  for (int m = 0; m < 5; ++m) {
    u64 ov = (u64)__shfl_xor((unsigned long long)best[m], 16, 32);
    if (hi == 0) top5_insert64(best, ov);
  }

  if (hi == 0) {
    int j = colbase + jn;
    int* fb = flat + batch * (KNN * n);      // k-major, per reference flatten
#pragma unroll
    for (int k = 0; k < KNN; ++k) fb[k * n + j] = (int)(unsigned)(best[k] & 0xFFFFFFFFu);
  }
}

// ---------------------------------------------------------------- logprobs (faithful k-major / node-major mismatch)
__global__ void k_logprob(const float* __restrict__ ge, const int* __restrict__ flat,
                          const float* __restrict__ tptr, float* __restrict__ dout, int layer) {
  int u = blockIdx.x * blockDim.x + threadIdx.x;
  if (u >= NB * NK) return;
  int b = u / NK, m = u % NK;
  float temp = fminf(fmaxf(tptr[0], -5.0f), 5.0f);
  float t = expf(temp);
  int nbr = flat[b * NK + m];                // x1: k-major gather
  int q = m / KNN, k2 = m % KNN;             // x2: node-major repeat
  const float4* x1 = (const float4*)(ge + ((size_t)b * NPB + nbr) * 64);
  const float4* x2 = (const float4*)(ge + ((size_t)b * NPB + q)   * 64);
  float s = 0.f;
#pragma unroll
  for (int i = 0; i < 16; ++i) {
    float4 aa = x1[i], bb = x2[i];
    float dx = aa.x - bb.x, dy = aa.y - bb.y, dz = aa.z - bb.z, dw = aa.w - bb.w;
    s += dx * dx + dy * dy + dz * dz + dw * dw;
  }
  dout[NTOT * 8 + (((size_t)b * NPB + q) * KNN + k2) * 2 + layer] = -s * t;
}

// ---------------------------------------------------------------- init max-accumulator to enc(-inf)
__global__ void k_inity(unsigned* __restrict__ y) {
  int i = blockIdx.x * blockDim.x + threadIdx.x;
  if (i < NTOT * 64) y[i] = 0x007FFFFFu;
}

// ---------------------------------------------------------------- EdgeConv with scrambled edge reconstruction
__global__ __launch_bounds__(64) void k_edgeconv(const float* __restrict__ xf,
                                                 const int* __restrict__ flat,
                                                 const float* __restrict__ W,
                                                 const float* __restrict__ b,
                                                 unsigned* __restrict__ y) {
  int s = blockIdx.x, t = threadIdx.x;
  int mm  = s >> 2;
  int bb  = (s >> 1) & 1;
  int c   = s & 1;
  int mm2 = mm + NEDGE / 4;                  // +10240
  const int* fb = flat + bb * NK;
  int src = (c == 0) ? (fb[mm]  + bb * NPB) : (mm  / KNN + bb * NPB);
  int dst = (c == 0) ? (fb[mm2] + bb * NPB) : (mm2 / KNN + bb * NPB);
  __shared__ float xi[64], xj[64];
  xi[t] = xf[(size_t)dst * 64 + t];
  xj[t] = xf[(size_t)src * 64 + t];
  __syncthreads();
  float acc = b[t];
#pragma unroll 4
  for (int cc = 0; cc < 64; ++cc) {
    acc += xi[cc] * W[cc * 64 + t];
    acc += (xj[cc] - xi[cc]) * W[(64 + cc) * 64 + t];
  }
  atomicMax(&y[(size_t)dst * 64 + t], enc_f32(acc));
}

// ---------------------------------------------------------------- decode + (isfinite->0) + relu
__global__ void k_final(const unsigned* __restrict__ y, float* __restrict__ xo) {
  int i = blockIdx.x * blockDim.x + threadIdx.x;
  if (i >= NTOT * 64) return;
  float f = dec_f32(y[i]);
  xo[i] = (f == f) ? fmaxf(f, 0.0f) : 0.0f;  // -inf (empty segment) -> 0, then relu
}

// ---------------------------------------------------------------- fc head: 64 -> 32 (leaky) -> 8
__global__ __launch_bounds__(32) void k_fc(const float* __restrict__ x3,
                                           const float* __restrict__ W1, const float* __restrict__ b1,
                                           const float* __restrict__ W2, const float* __restrict__ b2,
                                           float* __restrict__ out) {
  int row = blockIdx.x, t = threadIdx.x;
  __shared__ float sx[64];
  __shared__ float sh[32];
  sx[t]      = x3[(size_t)row * 64 + t];
  sx[32 + t] = x3[(size_t)row * 64 + 32 + t];
  __syncthreads();
  float acc = b1[t];
#pragma unroll 4
  for (int c = 0; c < 64; ++c) acc += sx[c] * W1[c * 32 + t];
  sh[t] = acc > 0.0f ? acc : 0.1f * acc;
  __syncthreads();
  if (t < 8) {
    float o = b2[t];
#pragma unroll 4
    for (int c = 0; c < 32; ++c) o += sh[c] * W2[c * 8 + t];
    out[(size_t)row * 8 + t] = o;
  }
}

// ---------------------------------------------------------------- launch
extern "C" void kernel_launch(void* const* d_in, const int* in_sizes, int n_in,
                              void* d_out, int out_size, void* d_ws, size_t ws_size,
                              hipStream_t stream) {
  (void)in_sizes; (void)n_in; (void)out_size; (void)ws_size;
  const float* x     = (const float*)d_in[0];
  const float* W_pre = (const float*)d_in[1];
  const float* b_pre = (const float*)d_in[2];
  const float* W_d1  = (const float*)d_in[3];
  const float* b_d1  = (const float*)d_in[4];
  const float* t1    = (const float*)d_in[5];
  const float* W_c1  = (const float*)d_in[6];
  const float* b_c1  = (const float*)d_in[7];
  const float* W_d2  = (const float*)d_in[8];
  const float* b_d2  = (const float*)d_in[9];
  const float* t2    = (const float*)d_in[10];
  const float* W_c2  = (const float*)d_in[11];
  const float* b_c2  = (const float*)d_in[12];
  const float* W_f1  = (const float*)d_in[13];
  const float* b_f1  = (const float*)d_in[14];
  const float* W_f2  = (const float*)d_in[15];
  const float* b_f2  = (const float*)d_in[16];
  float* out = (float*)d_out;

  char* ws = (char*)d_ws;
  float*    xf    = (float*)   (ws + 0);
  float*    ge1   = (float*)   (ws + 2097152);
  _Float16* geh1  = (_Float16*)(ws + 4194304);
  float*    sq1   = (float*)   (ws + 5242880);
  int*      flat1 = (int*)     (ws + 5275648);
  unsigned* y1    = (unsigned*)(ws + 5439488);
  float*    x2    = (float*)   (ws + 7536640);
  float*    ge2   = (float*)   (ws + 9633792);
  _Float16* geh2  = (_Float16*)(ws + 11730944);
  float*    sq2   = (float*)   (ws + 12779520);
  int*      flat2 = (int*)     (ws + 12812288);
  unsigned* y2    = (unsigned*)(ws + 12976128);
  float*    x3    = (float*)   (ws + 15073280);

  // pre_fc
  k_prefc<<<NTOT, 64, 0, stream>>>(x, W_pre, b_pre, xf);

  // ---- layer 1
  k_embed<<<NTOT, 64, 0, stream>>>(xf, nullptr, W_d1, b_d1, ge1, geh1);
  k_rowsq<<<(NTOT + 255) / 256, 256, 0, stream>>>(ge1, sq1);
  k_topk<<<dim3(NPB / 64, NB), 128, 0, stream>>>(geh1, sq1, flat1);
  k_logprob<<<(NB * NK + 255) / 256, 256, 0, stream>>>(ge1, flat1, t1, out, 0);
  k_inity<<<(NTOT * 64 + 255) / 256, 256, 0, stream>>>(y1);
  k_edgeconv<<<NEDGE, 64, 0, stream>>>(xf, flat1, W_c1, b_c1, y1);
  k_final<<<(NTOT * 64 + 255) / 256, 256, 0, stream>>>(y1, x2);

  // ---- layer 2 (gx = [ge1, x2])
  k_embed<<<NTOT, 64, 0, stream>>>(ge1, x2, W_d2, b_d2, ge2, geh2);
  k_rowsq<<<(NTOT + 255) / 256, 256, 0, stream>>>(ge2, sq2);
  k_topk<<<dim3(NPB / 64, NB), 128, 0, stream>>>(geh2, sq2, flat2);
  k_logprob<<<(NB * NK + 255) / 256, 256, 0, stream>>>(ge2, flat2, t2, out, 1);
  k_inity<<<(NTOT * 64 + 255) / 256, 256, 0, stream>>>(y2);
  k_edgeconv<<<NEDGE, 64, 0, stream>>>(x2, flat2, W_c2, b_c2, y2);
  k_final<<<(NTOT * 64 + 255) / 256, 256, 0, stream>>>(y2, x3);

  // fc head
  k_fc<<<NTOT, 32, 0, stream>>>(x3, W_f1, b_f1, W_f2, b_f2, out);
}
